// Decoder_30657476558898
// MI455X (gfx1250) — matrix operand order repaired and checked by reading, not compiled, not run on hardware
//
#include <hip/hip_runtime.h>
#include <hip/hip_bf16.h>
#include <math.h>

#define LATENT 128
#define VOCAB  32000
#define EMB    512
#define HID    1024
#define BATCH  64

typedef __attribute__((ext_vector_type(16))) __bf16    v16bf;
typedef __attribute__((ext_vector_type(8)))  float     v8f;
typedef __attribute__((ext_vector_type(4)))  unsigned  u32x4;
typedef __attribute__((ext_vector_type(4)))  unsigned  uint32x4;
typedef __attribute__((ext_vector_type(8)))  int       int32x8;
typedef __attribute__((ext_vector_type(4)))  int       int32x4;

union FragBF { v16bf v; u32x4 q[2]; };

__device__ __forceinline__ unsigned short f2bf(float f) {
  union { float f; unsigned u; } v; v.f = f;
  unsigned u = v.u;
  if ((u & 0x7fffffffu) > 0x7f800000u) return (unsigned short)((u >> 16) | 0x0040u); // NaN
  return (unsigned short)((u + 0x7fffu + ((u >> 16) & 1u)) >> 16);                    // RNE
}

// ---------------- TDM (Tensor Data Mover) helpers ------------------------------
__device__ __forceinline__ void tdm_issue(uint32x4 g0, int32x8 g1, int32x4 g2, int32x4 g3) {
#if __clang_major__ >= 23
  int32x8 g4 = {0, 0, 0, 0, 0, 0, 0, 0};
  __builtin_amdgcn_tensor_load_to_lds(g0, g1, g2, g3, g4, 0);
#else
  __builtin_amdgcn_tensor_load_to_lds(g0, g1, g2, g3, 0);
#endif
}

// Linear copy: n8 x 8-byte units, global -> LDS. (1-row tile, data_size=8B)
__device__ __forceinline__ void tdm_copy_1d(unsigned lds_addr, const void* gptr, unsigned n8) {
  unsigned long long ga = (unsigned long long)(size_t)gptr;
  uint32x4 g0;
  g0[0] = 1u;                                               // count=1
  g0[1] = lds_addr;                                         // D# lds_addr [63:32]
  g0[2] = (unsigned)(ga & 0xffffffffu);                     // global_addr lo
  g0[3] = (unsigned)((ga >> 32) & 0x01ffffffu) | 0x80000000u; // addr hi | type=2
  int32x8 g1;
  g1[0] = (int)(3u << 16);                                  // data_size = 8 bytes
  g1[1] = (int)((n8 & 0xffffu) << 16);                      // tensor_dim0 [79:48] lo
  g1[2] = (int)(((n8 >> 16) & 0xffffu) | (1u << 16));       // dim0 hi | tensor_dim1=1
  g1[3] = (int)((n8 & 0xffffu) << 16);                      // tile_dim0 [127:112]
  g1[4] = 1;                                                // tile_dim1 = 1
  g1[5] = (int)n8;                                          // tensor_dim0_stride lo
  g1[6] = 0;
  g1[7] = 0;
  int32x4 z = {0, 0, 0, 0};
  tdm_issue(g0, g1, z, z);
}

// Gather copy: 16 rows of row8 x 8-byte units each, selected by 16-bit row
// indices (D# gather mode), global base -> LDS (rows compacted sequentially).
__device__ __forceinline__ void tdm_gather16(unsigned lds_addr, const void* gbase,
                                             unsigned row8, unsigned nrows,
                                             const int* __restrict__ rows) {
  unsigned long long ga = (unsigned long long)(size_t)gbase;
  uint32x4 g0;
  g0[0] = 1u | (1u << 31);                                  // count=1 | gather_mode, 16-bit idx
  g0[1] = lds_addr;
  g0[2] = (unsigned)(ga & 0xffffffffu);
  g0[3] = (unsigned)((ga >> 32) & 0x01ffffffu) | 0x80000000u;
  int32x8 g1;
  g1[0] = (int)(3u << 16);                                  // data_size = 8 bytes
  g1[1] = (int)((row8 & 0xffffu) << 16);                    // tensor_dim0 lo
  g1[2] = (int)(((row8 >> 16) & 0xffffu) | ((nrows & 0xffffu) << 16)); // dim0 hi | dim1 lo
  g1[3] = (int)(((nrows >> 16) & 0xffffu) | ((row8 & 0xffffu) << 16)); // dim1 hi | tile_dim0
  g1[4] = 16;                                               // tile_dim1 = #indices
  g1[5] = (int)row8;                                        // tensor_dim0_stride
  g1[6] = 0;
  g1[7] = 0;
  int32x4 g2, g3;
  g2[0] = (rows[0]  & 0xffff) | ((rows[1]  & 0xffff) << 16);
  g2[1] = (rows[2]  & 0xffff) | ((rows[3]  & 0xffff) << 16);
  g2[2] = (rows[4]  & 0xffff) | ((rows[5]  & 0xffff) << 16);
  g2[3] = (rows[6]  & 0xffff) | ((rows[7]  & 0xffff) << 16);
  g3[0] = (rows[8]  & 0xffff) | ((rows[9]  & 0xffff) << 16);
  g3[1] = (rows[10] & 0xffff) | ((rows[11] & 0xffff) << 16);
  g3[2] = (rows[12] & 0xffff) | ((rows[13] & 0xffff) << 16);
  g3[3] = (rows[14] & 0xffff) | ((rows[15] & 0xffff) << 16);
  tdm_issue(g0, g1, g2, g3);
}

// ---------------- fragment loaders ---------------------------------------------
// A fragment from an LDS-staged 16-row tile (row stride ld elems, rows 0..15)
// lane m=lane&15, half=lane>>4 ; elems 0..7 -> K=kbase+half*8+e ; 8..15 -> +16
__device__ __forceinline__ v16bf load_a_lds(const unsigned short* base,
                                            int ld, int kbase, int lane) {
  const unsigned short* p = base + (lane & 15) * ld + kbase + ((lane >> 4) << 3);
  FragBF f;
  f.q[0] = *(const u32x4*)p;
  f.q[1] = *(const u32x4*)(p + 16);
  return f.v;
}

// B: 32x16 (KxN) from W[N][K] row-major => B[k][n] = W[nbase+n][kbase+k].
// lane n=lane&15, half=lane>>4 ; elems 0..7 -> K=kbase+half*8+e ; 8..15 -> +16
__device__ __forceinline__ v16bf load_b(const unsigned short* __restrict__ base,
                                        int ld, int nbase, int kbase, int lane) {
  int n = nbase + (lane & 15);
  const unsigned short* p = base + (size_t)n * ld + kbase + ((lane >> 4) << 3);
  FragBF f;
  f.q[0] = *(const u32x4*)p;
  f.q[1] = *(const u32x4*)(p + 16);
  return f.v;
}

#define WMMA_BF16(A, B, C) \
  __builtin_amdgcn_wmma_f32_16x16x32_bf16(false, (A), false, (B), (short)0, (C), false, false)

// ---------------- setup kernels ------------------------------------------------
__global__ void cvt_f32_bf16(const float* __restrict__ src,
                             unsigned short* __restrict__ dst, int n) {
  int i = blockIdx.x * blockDim.x + threadIdx.x;
  if (i < n) dst[i] = f2bf(src[i]);
}

// h0 = z @ lth_w^T + lth_b ; c = 0 ; tok = 0
__global__ void init_state(const float* __restrict__ z, const float* __restrict__ lth_w,
                           const float* __restrict__ lth_b,
                           unsigned short* __restrict__ h_bf, float* __restrict__ c_st,
                           int* __restrict__ tok) {
  int idx = blockIdx.x * blockDim.x + threadIdx.x;   // [0, BATCH*HID)
  int b = idx >> 10, j = idx & (HID - 1);
  const float* zr = z + b * LATENT;
  const float* wr = lth_w + (size_t)j * LATENT;
  float acc = lth_b[j];
#pragma unroll 8
  for (int k = 0; k < LATENT; ++k) acc += zr[k] * wr[k];
  h_bf[idx] = f2bf(acc);
  c_st[idx] = 0.f;
  if (j == 0) tok[b] = 0;
}

// ---------------- per-step: fused gates GEMM + LSTM cell -----------------------
// Block = 128 thr (4 waves), all share one mtile; waves cover 4 consecutive
// ntiles. TDM stages the gathered x tile (gather-mode embedding lookup) and the
// h tile into LDS; B fragments double-buffered in registers.
__global__ void __launch_bounds__(128) lstm_gates_step(
    const unsigned short* __restrict__ emb_bf, const unsigned short* __restrict__ wih_bf,
    const unsigned short* __restrict__ whh_bf, const float* __restrict__ b_ih,
    const float* __restrict__ b_hh, const unsigned short* __restrict__ h_in,
    unsigned short* __restrict__ h_out, float* __restrict__ c_st,
    const int* __restrict__ tok) {
  __shared__ __align__(16) unsigned short xT[16 * EMB];   // 16 KB: gathered embeddings
  __shared__ __align__(16) unsigned short hT[16 * HID];   // 32 KB: previous hidden rows

  int lane  = threadIdx.x & 31;
  int wave  = threadIdx.x >> 5;                 // 0..3
  int mtile = blockIdx.x >> 4;                  // 0..3  (64 blocks)
  int ntile = ((blockIdx.x & 15) << 2) + wave;  // 0..63
  int mbase = mtile << 4;
  int nbase = ntile << 4;

  // ---- TDM staging (wave 0 issues the DMAs; all waves sync on the barrier) ----
  if (wave == 0) {
    tdm_gather16((unsigned)(size_t)xT, emb_bf, (EMB * 2) / 8, VOCAB, tok + mbase);
    tdm_copy_1d((unsigned)(size_t)hT, h_in + (size_t)mbase * HID, (16 * HID * 2) / 8);
    __builtin_amdgcn_s_wait_tensorcnt(0);
  }
  __syncthreads();

  v8f a0 = {}, a1 = {}, a2 = {}, a3 = {};

  // ---- phase 1: x @ w_ih^T (K = EMB) ----
  {
    v16bf ac  = load_a_lds(xT, EMB, 0, lane);
    v16bf b0c = load_b(wih_bf, EMB, 0 * HID + nbase, 0, lane);
    v16bf b1c = load_b(wih_bf, EMB, 1 * HID + nbase, 0, lane);
    v16bf b2c = load_b(wih_bf, EMB, 2 * HID + nbase, 0, lane);
    v16bf b3c = load_b(wih_bf, EMB, 3 * HID + nbase, 0, lane);
    for (int k = 0; k < EMB; k += 32) {
      int kn = (k + 32) & (EMB - 1);               // wraps to 0 on last iter
      v16bf an  = load_a_lds(xT, EMB, kn, lane);
      v16bf b0n = load_b(wih_bf, EMB, 0 * HID + nbase, kn, lane);
      v16bf b1n = load_b(wih_bf, EMB, 1 * HID + nbase, kn, lane);
      v16bf b2n = load_b(wih_bf, EMB, 2 * HID + nbase, kn, lane);
      v16bf b3n = load_b(wih_bf, EMB, 3 * HID + nbase, kn, lane);
      a0 = WMMA_BF16(ac, b0c, a0);
      a1 = WMMA_BF16(ac, b1c, a1);
      a2 = WMMA_BF16(ac, b2c, a2);
      a3 = WMMA_BF16(ac, b3c, a3);
      ac = an; b0c = b0n; b1c = b1n; b2c = b2n; b3c = b3n;
    }
  }
  // ---- phase 2: h @ w_hh^T (K = HID) ----
  {
    v16bf ac  = load_a_lds(hT, HID, 0, lane);
    v16bf b0c = load_b(whh_bf, HID, 0 * HID + nbase, 0, lane);
    v16bf b1c = load_b(whh_bf, HID, 1 * HID + nbase, 0, lane);
    v16bf b2c = load_b(whh_bf, HID, 2 * HID + nbase, 0, lane);
    v16bf b3c = load_b(whh_bf, HID, 3 * HID + nbase, 0, lane);
    for (int k = 0; k < HID; k += 32) {
      int kn = (k + 32) & (HID - 1);
      v16bf an  = load_a_lds(hT, HID, kn, lane);
      v16bf b0n = load_b(whh_bf, HID, 0 * HID + nbase, kn, lane);
      v16bf b1n = load_b(whh_bf, HID, 1 * HID + nbase, kn, lane);
      v16bf b2n = load_b(whh_bf, HID, 2 * HID + nbase, kn, lane);
      v16bf b3n = load_b(whh_bf, HID, 3 * HID + nbase, kn, lane);
      a0 = WMMA_BF16(ac, b0c, a0);
      a1 = WMMA_BF16(ac, b1c, a1);
      a2 = WMMA_BF16(ac, b2c, a2);
      a3 = WMMA_BF16(ac, b3c, a3);
      ac = an; b0c = b0n; b1c = b1n; b2c = b2n; b3c = b3n;
    }
  }

  // ---- LSTM cell epilogue ----
  int col = nbase + (lane & 15);
  float bi = b_ih[0 * HID + col] + b_hh[0 * HID + col];
  float bf = b_ih[1 * HID + col] + b_hh[1 * HID + col];
  float bg = b_ih[2 * HID + col] + b_hh[2 * HID + col];
  float bo = b_ih[3 * HID + col] + b_hh[3 * HID + col];
  int rbase = mbase + ((lane >> 4) << 3);

#pragma unroll
  for (int r = 0; r < 8; ++r) {
    int idx = (rbase + r) * HID + col;
    float iv = 1.f / (1.f + expf(-(a0[r] + bi)));
    float fv = 1.f / (1.f + expf(-(a1[r] + bf)));
    float gv = tanhf(a2[r] + bg);
    float ov = 1.f / (1.f + expf(-(a3[r] + bo)));
    float cv = fv * c_st[idx] + iv * gv;
    c_st[idx]  = cv;
    h_out[idx] = f2bf(ov * tanhf(cv));
  }
}

// ---------------- per-step: logits GEMM [64,1024]x[1024,32000] -----------------
// grid (250, 4): blockIdx.y = mtile; 8 waves/block cover 8 ntiles (128 cols).
// h tile staged via TDM; B double-buffered from L2 (fc_w bf16 is L2-resident).
__global__ void __launch_bounds__(256) logits_step(
    const unsigned short* __restrict__ fcw_bf, const float* __restrict__ fc_b,
    const unsigned short* __restrict__ h_bf, float* __restrict__ out,
    int t, int seq) {
  __shared__ __align__(16) unsigned short hT[16 * HID];   // 32 KB

  int lane  = threadIdx.x & 31;
  int mtile = blockIdx.y;
  int ntile = blockIdx.x * 8 + (threadIdx.x >> 5);
  int mbase = mtile << 4;
  int nbase = ntile << 4;

  if ((threadIdx.x >> 5) == 0) {
    tdm_copy_1d((unsigned)(size_t)hT, h_bf + (size_t)mbase * HID, (16 * HID * 2) / 8);
    __builtin_amdgcn_s_wait_tensorcnt(0);
  }
  __syncthreads();

  v8f acc = {};
  v16bf ac = load_a_lds(hT, HID, 0, lane);
  v16bf bc = load_b(fcw_bf, HID, nbase, 0, lane);
  for (int k = 0; k < HID; k += 32) {
    int kn = (k + 32) & (HID - 1);
    v16bf an = load_a_lds(hT, HID, kn, lane);
    v16bf bn = load_b(fcw_bf, HID, nbase, kn, lane);
    acc = WMMA_BF16(ac, bc, acc);
    ac = an; bc = bn;
  }

  int col = nbase + (lane & 15);
  float bias = fc_b[col];
  int rbase = mbase + ((lane >> 4) << 3);
  size_t stepoff = (size_t)t * VOCAB;
#pragma unroll
  for (int r = 0; r < 8; ++r) {
    size_t row = (size_t)(rbase + r);
    out[row * seq * VOCAB + stepoff + col] = acc[r] + bias;
  }
}

// ---------------- per-step: greedy argmax per batch row ------------------------
__global__ void __launch_bounds__(256) argmax_step(const float* __restrict__ out,
                                                   int* __restrict__ tok,
                                                   int t, int seq) {
  int b = blockIdx.x;
  const float* row = out + (size_t)b * seq * VOCAB + (size_t)t * VOCAB;
  __shared__ float sv[256];
  __shared__ int   si[256];
  float best = -3.402823466e38f;
  int   bi   = VOCAB;
  for (int v = threadIdx.x; v < VOCAB; v += 256) {
    float x = row[v];
    if (x > best || (x == best && v < bi)) { best = x; bi = v; }
  }
  sv[threadIdx.x] = best; si[threadIdx.x] = bi;
  __syncthreads();
  for (int s = 128; s > 0; s >>= 1) {
    if (threadIdx.x < s) {
      float xo = sv[threadIdx.x + s]; int io = si[threadIdx.x + s];
      if (xo > sv[threadIdx.x] || (xo == sv[threadIdx.x] && io < si[threadIdx.x])) {
        sv[threadIdx.x] = xo; si[threadIdx.x] = io;
      }
    }
    __syncthreads();
  }
  if (threadIdx.x == 0) tok[b] = si[0];
}

// ---------------- host driver --------------------------------------------------
extern "C" void kernel_launch(void* const* d_in, const int* in_sizes, int n_in,
                              void* d_out, int out_size, void* d_ws, size_t ws_size,
                              hipStream_t stream) {
  (void)in_sizes; (void)n_in; (void)ws_size;
  const float* z     = (const float*)d_in[0];
  const float* emb   = (const float*)d_in[1];
  const float* w_ih  = (const float*)d_in[2];
  const float* w_hh  = (const float*)d_in[3];
  const float* b_ih  = (const float*)d_in[4];
  const float* b_hh  = (const float*)d_in[5];
  const float* fc_w  = (const float*)d_in[6];
  const float* fc_b  = (const float*)d_in[7];
  const float* lth_w = (const float*)d_in[8];
  const float* lth_b = (const float*)d_in[9];
  float* out = (float*)d_out;
  int seq = out_size / (BATCH * VOCAB);   // seq_len lives on device; recover from shape

  char* ws = (char*)d_ws;
  size_t off = 0;
  auto carve = [&](size_t bytes) {
    char* p = ws + off;
    off += (bytes + 255) & ~(size_t)255;
    return p;
  };
  unsigned short* fcw_bf = (unsigned short*)carve((size_t)VOCAB * HID * 2);   // 65.5 MB (L2-resident)
  unsigned short* wih_bf = (unsigned short*)carve((size_t)4 * HID * EMB * 2); //  4.2 MB
  unsigned short* whh_bf = (unsigned short*)carve((size_t)4 * HID * HID * 2); //  8.4 MB
  unsigned short* emb_bf = (unsigned short*)carve((size_t)VOCAB * EMB * 2);   // 32.8 MB
  unsigned short* h0_bf  = (unsigned short*)carve((size_t)BATCH * HID * 2);
  unsigned short* h1_bf  = (unsigned short*)carve((size_t)BATCH * HID * 2);
  float*          c_st   = (float*)carve((size_t)BATCH * HID * 4);
  int*            tok    = (int*)carve((size_t)BATCH * 4);

  {
    int n;
    n = VOCAB * HID;   cvt_f32_bf16<<<(n + 255) / 256, 256, 0, stream>>>(fc_w, fcw_bf, n);
    n = 4 * HID * EMB; cvt_f32_bf16<<<(n + 255) / 256, 256, 0, stream>>>(w_ih, wih_bf, n);
    n = 4 * HID * HID; cvt_f32_bf16<<<(n + 255) / 256, 256, 0, stream>>>(w_hh, whh_bf, n);
    n = VOCAB * EMB;   cvt_f32_bf16<<<(n + 255) / 256, 256, 0, stream>>>(emb, emb_bf, n);
  }
  init_state<<<(BATCH * HID) / 256, 256, 0, stream>>>(z, lth_w, lth_b, h0_bf, c_st, tok);

  unsigned short* hin  = h0_bf;
  unsigned short* hout = h1_bf;
  for (int t = 0; t < seq; ++t) {
    lstm_gates_step<<<64, 128, 0, stream>>>(emb_bf, wih_bf, whh_bf, b_ih, b_hh,
                                            hin, hout, c_st, tok);
    logits_step<<<dim3(VOCAB / 128, BATCH / 16), 256, 0, stream>>>(fcw_bf, fc_b, hout,
                                                                   out, t, seq);
    argmax_step<<<BATCH, 256, 0, stream>>>(out, tok, t, seq);
    unsigned short* tmp = hin; hin = hout; hout = tmp;
  }
}
